// MultiHeadSelfAttentionFormal_10393820857174
// MI455X (gfx1250) — compile-verified, matrix operand order checked by reading, not run on hardware
//
#include <hip/hip_runtime.h>

typedef __bf16 bf16;
typedef __attribute__((ext_vector_type(16))) __bf16 bf16x16;
typedef __attribute__((ext_vector_type(8)))  __bf16 bf16x8;
typedef __attribute__((ext_vector_type(4)))  __bf16 bf16x4;
typedef __attribute__((ext_vector_type(8)))  float  f32x8;
typedef int v4i __attribute__((vector_size(16)));

#define NB   2
#define NS   2048
#define ND   1024
#define NH   16
#define NDH  64
#define NTOK (NB * NS)

// ---------------------------------------------------------------------------
// CDNA5 async global->LDS staging (ASYNCcnt-tracked DMA, bypasses VGPRs).
// Builtin signature (from hipcc diagnostic): (v4i as1*, v4i as3*, imm, imm).
// Falls back to plain load/store staging if the builtins are not declared.
// ---------------------------------------------------------------------------
#if __has_builtin(__builtin_amdgcn_global_load_async_to_lds_b128) && \
    __has_builtin(__builtin_amdgcn_s_wait_asynccnt)
#define HAVE_ASYNC_LDS 1
typedef __attribute__((address_space(1))) v4i gv4i;
typedef __attribute__((address_space(3))) v4i lv4i;
#endif

static __device__ __forceinline__ void async_cp16(bf16* ldst, const bf16* gsrc) {
#if defined(HAVE_ASYNC_LDS)
  __builtin_amdgcn_global_load_async_to_lds_b128(
      (gv4i*)gsrc, (lv4i*)ldst, /*offset=*/0, /*cpol=*/0);
#else
  *(bf16x8*)ldst = *(const bf16x8*)gsrc;
#endif
}

template <int N>
static __device__ __forceinline__ void wait_async() {
#if defined(HAVE_ASYNC_LDS)
  __builtin_amdgcn_s_wait_asynccnt(N);
#endif
}

// ---------------------------------------------------------------------------
// WMMA fragment helpers (CDNA5 ISA §7.12.2, wave32 layouts).
// A 16x32 bf16 fragment: lane L (L<16) holds row L; halves split the K range:
//   hf=0 -> k 0..7 & 16..23, hf=1 -> k 8..15 & 24..31  (two 16B loads/lane).
// B 32x16 uses the mirrored layout: lane = output column, same K packing, so
// loading rows of the *transposed* B source with identical math is correct.
// ---------------------------------------------------------------------------
static __device__ __forceinline__ bf16x16 load_frag(const bf16* rowp, int hf) {
  union { bf16x16 v; bf16x8 h[2]; } f;
  f.h[0] = *(const bf16x8*)(rowp + hf * 8);
  f.h[1] = *(const bf16x8*)(rowp + 16 + hf * 8);
  return f.v;
}

static __device__ __forceinline__ f32x8 wmma_bf16(bf16x16 a, bf16x16 b, f32x8 c) {
  return __builtin_amdgcn_wmma_f32_16x16x32_bf16(
      /*neg_a=*/false, a, /*neg_b=*/false, b,
      /*c_mod=*/(short)0, c, /*reuse_a=*/false, /*reuse_b=*/false);
}

static __device__ __forceinline__ float redmax16(float v) {
  v = fmaxf(v, __shfl_xor(v, 1));
  v = fmaxf(v, __shfl_xor(v, 2));
  v = fmaxf(v, __shfl_xor(v, 4));
  v = fmaxf(v, __shfl_xor(v, 8));
  return v;  // stays within each 16-lane half (rows differ per half)
}
static __device__ __forceinline__ float redsum16(float v) {
  v += __shfl_xor(v, 1);
  v += __shfl_xor(v, 2);
  v += __shfl_xor(v, 4);
  v += __shfl_xor(v, 8);
  return v;
}

// ---------------------------------------------------------------------------
// f32 -> bf16 conversion (vectorized x4)
// ---------------------------------------------------------------------------
__global__ __launch_bounds__(256)
void cvt_f32_bf16(const float* __restrict__ in, bf16* __restrict__ out, int n4) {
  int i = blockIdx.x * 256 + threadIdx.x;
  if (i < n4) {
    float4 v = ((const float4*)in)[i];
    bf16x4 o;
    o.x = (bf16)v.x; o.y = (bf16)v.y; o.z = (bf16)v.z; o.w = (bf16)v.w;
    ((bf16x4*)out)[i] = o;
  }
}

// ---------------------------------------------------------------------------
// GEMM:  Y[m,n] = sum_k A[m,k] * W[n,k] + bias[n]      (A @ W^T + b)
// Block tile 128x128, 8 waves (4 M x 2 N), wave tile 32x64.
// K staged 64 at a time into double-buffered LDS via async-to-LDS DMA;
// next tile streams in while the current one is consumed by WMMAs.
// MODE 0: out bf16 [B,H,S,DH]   (Q / K head layout)
// MODE 1: out bf16 [B,H,DH,S]   (V transposed for the PV matmul)
// MODE 2: out f32  [M x 1024]   (final projection)
// ---------------------------------------------------------------------------
template <int MODE>
__global__ __launch_bounds__(256)
void gemm_bf16(const bf16* __restrict__ A, const bf16* __restrict__ W,
               const float* __restrict__ bias, void* __restrict__ out) {
  constexpr int K   = ND;
  constexpr int BK  = 64;
  constexpr int KT  = K / BK;  // 16 stages
  constexpr int LDT = 72;      // 64 + 8 pad (144B rows, 16B aligned)
  __shared__ bf16 sA[2][128 * LDT];
  __shared__ bf16 sB[2][128 * LDT];

  const int tid  = threadIdx.x;
  const int lane = tid & 31;
  const int wave = tid >> 5;
  const int r    = lane & 15;
  const int hf   = lane >> 4;
  const int wm   = (wave & 3) * 32;
  const int wn   = (wave >> 2) * 64;
  const int bm   = blockIdx.x * 128;
  const int bn   = blockIdx.y * 128;

  // Stage one 128x64 tile of A and of W: 1024 16B chunks each, 4 per thread.
  auto stage = [&](int buf, int k0) {
#pragma unroll
    for (int p = 0; p < 4; ++p) {
      int c   = tid + p * 256;
      int row = c >> 3;
      int ch  = (c & 7) * 8;
      async_cp16(&sA[buf][row * LDT + ch], &A[(size_t)(bm + row) * K + k0 + ch]);
      async_cp16(&sB[buf][row * LDT + ch], &W[(size_t)(bn + row) * K + k0 + ch]);
    }
  };

  f32x8 acc[2][4];
#pragma unroll
  for (int i = 0; i < 2; ++i)
#pragma unroll
    for (int j = 0; j < 4; ++j)
#pragma unroll
      for (int e = 0; e < 8; ++e) acc[i][j][e] = 0.0f;

  stage(0, 0);
  for (int kt = 0; kt < KT; ++kt) {
    const int buf = kt & 1;
    if (kt + 1 < KT) {
      stage(buf ^ 1, (kt + 1) * BK);  // overlap next-tile DMA with compute
      wait_async<8>();                // first 8 asyncs (= current tile) done
    } else {
      wait_async<0>();
    }
    __syncthreads();

#pragma unroll
    for (int kf = 0; kf < 2; ++kf) {  // two 32-wide K fragments per stage
      bf16x16 fa[2], fb[4];
#pragma unroll
      for (int mi = 0; mi < 2; ++mi)
        fa[mi] = load_frag(&sA[buf][(wm + mi * 16 + r) * LDT + kf * 32], hf);
#pragma unroll
      for (int ni = 0; ni < 4; ++ni)
        fb[ni] = load_frag(&sB[buf][(wn + ni * 16 + r) * LDT + kf * 32], hf);
#pragma unroll
      for (int mi = 0; mi < 2; ++mi)
#pragma unroll
        for (int ni = 0; ni < 4; ++ni)
          acc[mi][ni] = wmma_bf16(fa[mi], fb[ni], acc[mi][ni]);
    }
    __syncthreads();
  }

  // Epilogue. C layout: VGPR e -> row e (lanes 0-15) / e+8 (lanes 16-31),
  // column = lane & 15.
#pragma unroll
  for (int mi = 0; mi < 2; ++mi) {
#pragma unroll
    for (int ni = 0; ni < 4; ++ni) {
      int   n  = bn + wn + ni * 16 + r;
      float bv = bias[n];
#pragma unroll
      for (int e = 0; e < 8; ++e) {
        int   m   = bm + wm + mi * 16 + e + hf * 8;
        float val = acc[mi][ni][e] + bv;
        if (MODE == 2) {
          ((float*)out)[(size_t)m * ND + n] = val;
        } else {
          int b = m >> 11, s = m & (NS - 1);
          int h = n >> 6,  d = n & (NDH - 1);
          if (MODE == 0)
            ((bf16*)out)[(((size_t)(b * NH + h)) * NS + s) * NDH + d] = (bf16)val;
          else
            ((bf16*)out)[(((size_t)(b * NH + h)) * NDH + d) * NS + s] = (bf16)val;
        }
      }
    }
  }
}

// ---------------------------------------------------------------------------
// Flash attention: one wave owns a 16-query tile, block covers 128 queries.
// 64-key K/V tiles double-buffered in LDS via async-to-LDS DMA so the next
// tile streams in during softmax + PV WMMAs. Online softmax (running row
// max m, row sumexp l). P is round-tripped through a per-wave LDS buffer to
// convert the C-fragment layout into an A-fragment.
// Qh/Kh: [B*H][S][DH] bf16.  Vt: [B*H][DH][S] bf16.  Ot: [NTOK][ND] bf16.
// ---------------------------------------------------------------------------
__global__ __launch_bounds__(256)
void attention(const bf16* __restrict__ Qh, const bf16* __restrict__ Kh,
               const bf16* __restrict__ Vt, bf16* __restrict__ Ot) {
  constexpr int LK = 72;  // 64 + 8 pad (144B rows)
  constexpr int KT = NS / 64;
  __shared__ bf16 sK[2][64 * LK];    // [key 0..63][dh 0..63]
  __shared__ bf16 sV[2][64 * LK];    // [dh 0..63][key 0..63]
  __shared__ bf16 sP[8][16 * LK];    // per-wave 16x64 probability tile

  const int tid  = threadIdx.x;
  const int lane = tid & 31;
  const int wave = tid >> 5;
  const int r    = lane & 15;
  const int hf   = lane >> 4;
  const int bh   = blockIdx.x;                    // 0..31 = b*NH + h
  const int q0   = blockIdx.y * 128 + wave * 16;  // query tile base

  auto stage = [&](int buf, int k0) {
#pragma unroll
    for (int p = 0; p < 2; ++p) {
      int c   = tid + p * 256;  // 512 chunks of 8 bf16 per 64x64 tile
      int row = c >> 3;
      int ch  = (c & 7) * 8;
      async_cp16(&sK[buf][row * LK + ch],
                 &Kh[((size_t)bh * NS + k0 + row) * NDH + ch]);
      async_cp16(&sV[buf][row * LK + ch],
                 &Vt[((size_t)bh * NDH + row) * NS + k0 + ch]);
    }
  };

  // Q fragments for this wave's 16 rows (k = dh 0..31 and 32..63)
  const bf16* qrow = Qh + ((size_t)bh * NS + q0 + r) * NDH;
  bf16x16 aq0 = load_frag(qrow, hf);
  bf16x16 aq1 = load_frag(qrow + 32, hf);

  f32x8 accO[4];
#pragma unroll
  for (int i = 0; i < 4; ++i)
#pragma unroll
    for (int e = 0; e < 8; ++e) accO[i][e] = 0.0f;
  float mrow[8], lrow[8];
#pragma unroll
  for (int e = 0; e < 8; ++e) { mrow[e] = -1e30f; lrow[e] = 0.0f; }

  stage(0, 0);
  for (int kt = 0; kt < KT; ++kt) {
    const int buf = kt & 1;
    if (kt + 1 < KT) {
      stage(buf ^ 1, (kt + 1) * 64);
      wait_async<4>();
    } else {
      wait_async<0>();
    }
    __syncthreads();

    // ---- scores S = (Q K^T) * 1/sqrt(DH) for 64 keys (4 x 16-wide tiles)
    float sc[4][8];
#pragma unroll
    for (int ni = 0; ni < 4; ++ni) {
      bf16x16 bk0 = load_frag(&sK[buf][(ni * 16 + r) * LK], hf);       // dh 0..31
      bf16x16 bk1 = load_frag(&sK[buf][(ni * 16 + r) * LK + 32], hf);  // dh 32..63
      f32x8 a;
#pragma unroll
      for (int e = 0; e < 8; ++e) a[e] = 0.0f;
      a = wmma_bf16(aq0, bk0, a);
      a = wmma_bf16(aq1, bk1, a);
#pragma unroll
      for (int e = 0; e < 8; ++e) sc[ni][e] = a[e] * 0.125f;  // 1/sqrt(64)
    }

    // ---- online softmax update
#pragma unroll
    for (int e = 0; e < 8; ++e) {
      float t = sc[0][e];
      t = fmaxf(t, sc[1][e]); t = fmaxf(t, sc[2][e]); t = fmaxf(t, sc[3][e]);
      t = redmax16(t);
      float mnew = fmaxf(mrow[e], t);
      float corr = __expf(mrow[e] - mnew);
      mrow[e] = mnew;
      lrow[e] *= corr;
#pragma unroll
      for (int ni = 0; ni < 4; ++ni) accO[ni][e] *= corr;
      float ps = 0.0f;
#pragma unroll
      for (int ni = 0; ni < 4; ++ni) {
        float pv = __expf(sc[ni][e] - mnew);
        ps += pv;
        sP[wave][(e + hf * 8) * LK + ni * 16 + r] = (bf16)pv;
      }
      lrow[e] += redsum16(ps);
    }

    // ---- O += P @ V  (same-wave LDS RAW is in order; compiler waits dscnt)
    bf16x16 ap0 = load_frag(&sP[wave][r * LK], hf);       // keys 0..31
    bf16x16 ap1 = load_frag(&sP[wave][r * LK + 32], hf);  // keys 32..63
#pragma unroll
    for (int ni = 0; ni < 4; ++ni) {
      bf16x16 bv0 = load_frag(&sV[buf][(ni * 16 + r) * LK], hf);
      bf16x16 bv1 = load_frag(&sV[buf][(ni * 16 + r) * LK + 32], hf);
      accO[ni] = wmma_bf16(ap0, bv0, accO[ni]);
      accO[ni] = wmma_bf16(ap1, bv1, accO[ni]);
    }
    __syncthreads();
  }

  // ---- normalize and write token-major bf16 for the output projection
  const int b = bh >> 4, h = bh & 15;
#pragma unroll
  for (int ni = 0; ni < 4; ++ni) {
#pragma unroll
    for (int e = 0; e < 8; ++e) {
      int    sq  = q0 + e + hf * 8;
      int    d   = ni * 16 + r;
      size_t tok = (size_t)b * NS + sq;
      Ot[tok * ND + h * NDH + d] = (bf16)(accO[ni][e] / lrow[e]);
    }
  }
}

// ---------------------------------------------------------------------------
// Host launcher
// ---------------------------------------------------------------------------
extern "C" void kernel_launch(void* const* d_in, const int* in_sizes, int n_in,
                              void* d_out, int out_size, void* d_ws, size_t ws_size,
                              hipStream_t stream) {
  const float* X  = (const float*)d_in[0];
  const float* Wq = (const float*)d_in[1];
  const float* bq = (const float*)d_in[2];
  const float* Wk = (const float*)d_in[3];
  const float* bk = (const float*)d_in[4];
  const float* Wv = (const float*)d_in[5];
  const float* bv = (const float*)d_in[6];
  const float* Wo = (const float*)d_in[7];
  const float* bo = (const float*)d_in[8];

  char* ws = (char*)d_ws;
  bf16* Xb  = (bf16*)ws; ws += (size_t)NTOK * ND * 2;
  bf16* Wqb = (bf16*)ws; ws += (size_t)ND * ND * 2;
  bf16* Wkb = (bf16*)ws; ws += (size_t)ND * ND * 2;
  bf16* Wvb = (bf16*)ws; ws += (size_t)ND * ND * 2;
  bf16* Wob = (bf16*)ws; ws += (size_t)ND * ND * 2;
  bf16* Qh  = (bf16*)ws; ws += (size_t)NB * NH * NS * NDH * 2;
  bf16* Kh  = (bf16*)ws; ws += (size_t)NB * NH * NS * NDH * 2;
  bf16* Vt  = (bf16*)ws; ws += (size_t)NB * NH * NDH * NS * 2;
  bf16* Ot  = (bf16*)ws; ws += (size_t)NTOK * ND * 2;

  const int nX = NTOK * ND / 4, nW = ND * ND / 4;
  cvt_f32_bf16<<<(nX + 255) / 256, 256, 0, stream>>>(X,  Xb,  nX);
  cvt_f32_bf16<<<(nW + 255) / 256, 256, 0, stream>>>(Wq, Wqb, nW);
  cvt_f32_bf16<<<(nW + 255) / 256, 256, 0, stream>>>(Wk, Wkb, nW);
  cvt_f32_bf16<<<(nW + 255) / 256, 256, 0, stream>>>(Wv, Wvb, nW);
  cvt_f32_bf16<<<(nW + 255) / 256, 256, 0, stream>>>(Wo, Wob, nW);

  dim3 blk(256);
  dim3 gg(NTOK / 128, ND / 128);
  gemm_bf16<0><<<gg, blk, 0, stream>>>(Xb, Wqb, bq, Qh);
  gemm_bf16<0><<<gg, blk, 0, stream>>>(Xb, Wkb, bk, Kh);
  gemm_bf16<1><<<gg, blk, 0, stream>>>(Xb, Wvb, bv, Vt);

  attention<<<dim3(NB * NH, NS / 128), blk, 0, stream>>>(Qh, Kh, Vt, Ot);

  gemm_bf16<2><<<gg, blk, 0, stream>>>(Ot, Wob, bo, d_out);
}
